// DeformedConvModel_51436528337590
// MI455X (gfx1250) — matrix operand order repaired and checked by reading, not compiled, not run on hardware
//
#include <hip/hip_runtime.h>
#include <hip/hip_bf16.h>

typedef __attribute__((ext_vector_type(16))) _Float16 v16h;
typedef __attribute__((ext_vector_type(8)))  float    v8f;

#define NIMG 2048

// ---------------------------------------------------------------- utilities
__global__ void zero_kernel(float* p, int n) {
  int i = blockIdx.x * blockDim.x + threadIdx.x;
  if (i < n) p[i] = 0.0f;
}

// Convert f32 OIHW weights (Cout, Cin*9) into WMMA A-fragment order:
// dst[((mt*NKT + kt)*32 + lane)*16 + h], lane/h per the 16-bit A-matrix VGPR
// layout (lanes 0-15: K%32 in {0..7,16..23}; lanes 16-31: {8..15,24..31}).
__global__ void cvt_frag_kernel(const float* __restrict__ src, _Float16* __restrict__ dst,
                                int KDIM, int n) {
  int idx = blockIdx.x * blockDim.x + threadIdx.x;
  if (idx >= n) return;
  int m = idx / KDIM;
  int K = idx - m * KDIM;
  int kt = K >> 5, kw = K & 31;
  int lhalf = (kw >> 3) & 1;
  int h = (kw & 7) + ((kw & 16) ? 8 : 0);
  int lane = (m & 15) + lhalf * 16;
  int mt = m >> 4;
  int NKT = KDIM >> 5;
  dst[(size_t)(((mt * NKT + kt) << 5) + lane) * 16 + h] = (_Float16)src[idx];
}

// scale = g * rsqrt(var+eps); shift = b - mean*scale
__global__ void bn_finalize_kernel(const float* __restrict__ sum, const float* __restrict__ sumsq,
                                   const float* __restrict__ g, const float* __restrict__ bb,
                                   float* __restrict__ scale, float* __restrict__ shift,
                                   float inv_count) {
  int c = threadIdx.x;
  float m = sum[c] * inv_count;
  float v = sumsq[c] * inv_count - m * m;
  float s = g[c] * rsqrtf(v + 1e-5f);
  scale[c] = s;
  shift[c] = bb[c] - m * s;
}

// ------------------------------------------------- conv1 (1->32, 3x3, valid) + relu + bn1 stats
__global__ void __launch_bounds__(256) conv1_kernel(const float* __restrict__ x,
                                                    const float* __restrict__ w,
                                                    const float* __restrict__ bias,
                                                    float* __restrict__ h1,
                                                    float* __restrict__ csum,
                                                    float* __restrict__ csumsq) {
  extern __shared__ char smraw[];
  float* xs = (float*)smraw;          // 18*18
  float* cs = xs + 18 * 18;           // 32
  float* cq = cs + 32;                // 32
  const int b = blockIdx.x, tid = threadIdx.x;
  for (int i = tid; i < 18 * 18; i += 256) xs[i] = x[(size_t)b * 324 + i];
  if (tid < 32) { cs[tid] = 0.0f; cq[tid] = 0.0f; }
  __syncthreads();
  const int ho = tid >> 4, wo = tid & 15;
  float win[9];
#pragma unroll
  for (int ki = 0; ki < 3; ++ki)
#pragma unroll
    for (int kj = 0; kj < 3; ++kj) win[ki * 3 + kj] = xs[(ho + ki) * 18 + (wo + kj)];
  for (int c = 0; c < 32; ++c) {
    float acc = bias[c];
#pragma unroll
    for (int i = 0; i < 9; ++i) acc += win[i] * w[c * 9 + i];
    acc = fmaxf(acc, 0.0f);
    h1[((size_t)b * 32 + c) * 256 + tid] = acc;
    atomicAdd(&cs[c], acc);
    atomicAdd(&cq[c], acc * acc);
  }
  __syncthreads();
  if (tid < 32) {
    atomicAdd(&csum[tid], cs[tid]);
    atomicAdd(&csumsq[tid], cq[tid]);
  }
}

// ------------------------------------------------- offset conv (Cin -> 18, 3x3, pad 1) direct
template <int CIN, int HIN, int WIN, int HO, int WO, int STRIDE>
__global__ void __launch_bounds__(128) off_conv_kernel(const float* __restrict__ hin,
                                                       const float* __restrict__ sc,
                                                       const float* __restrict__ sh,
                                                       const float* __restrict__ w,
                                                       const float* __restrict__ bias,
                                                       float* __restrict__ out) {
  extern __shared__ char smraw[];
  float* hs = (float*)smraw;  // CIN*HIN*WIN, bn-applied
  const int b = blockIdx.x, tid = threadIdx.x;
  const float* hb = hin + (size_t)b * CIN * (HIN * WIN);
  for (int i = tid; i < CIN * HIN * WIN; i += 128) {
    int c = i / (HIN * WIN);
    hs[i] = hb[i] * sc[c] + sh[c];
  }
  __syncthreads();
  const int NOUT = 18 * HO * WO;
  for (int o = tid; o < NOUT; o += 128) {
    int co = o / (HO * WO);
    int pix = o - co * (HO * WO);
    int ho = pix / WO, wo = pix - ho * WO;
    float acc = bias[co];
    for (int ci = 0; ci < CIN; ++ci) {
      const float* hc = hs + ci * (HIN * WIN);
      const float* wc = w + ((size_t)co * CIN + ci) * 9;
#pragma unroll
      for (int ki = 0; ki < 3; ++ki) {
        int y = ho * STRIDE - 1 + ki;
        if ((unsigned)y >= (unsigned)HIN) continue;
#pragma unroll
        for (int kj = 0; kj < 3; ++kj) {
          int xx = wo * STRIDE - 1 + kj;
          if ((unsigned)xx >= (unsigned)WIN) continue;
          acc += hc[y * WIN + xx] * wc[ki * 3 + kj];
        }
      }
    }
    out[((size_t)b * 18 + co) * (HO * WO) + pix] = acc;
  }
}

// ------------------------------------------------- deformable conv via WMMA
// One block (4 waves, 128 threads) per image.
// Phase 0: stage bn(h) input tile into LDS (f32)
// Phase 1: bilinear-gather from LDS, scatter f16 samples DIRECTLY into
//          B-fragment order: frag[((kt*NT + nt)*32 + lane)*16 + h]
// Phase 2: GEMM out[Cout,Npix] = W @ val via v_wmma_f32_16x16x32_f16 with
//          contiguous 32B v16h loads (ds_load_b128 / global_load_b128)
// Fused:   relu + per-channel sum/sumsq (shfl reduce + atomics) for next BN.
template <int CIN, int COUT, int HIN, int WIN, int HO, int WO, int STRIDE>
__global__ void __launch_bounds__(128) deform_wmma_kernel(const float* __restrict__ hin,
                                                          const float* __restrict__ sc,
                                                          const float* __restrict__ sh,
                                                          const float* __restrict__ off,
                                                          const _Float16* __restrict__ wfrag,
                                                          float* __restrict__ hout,
                                                          float* __restrict__ osum,
                                                          float* __restrict__ osumsq) {
  constexpr int NPIX = HO * WO;       // 64 or 16
  constexpr int KDIM = CIN * 9;       // 288 / 576 / 1152
  constexpr int NT = NPIX / 16;       // N tiles
  constexpr int NKT = KDIM / 32;      // K tiles
  constexpr int NMT = COUT / 16;      // M tiles
  constexpr int INSZ = CIN * HIN * WIN;
  extern __shared__ char smraw[];
  _Float16* frag = (_Float16*)smraw;                      // KDIM*NPIX halves, fragment order
  float* hs = (float*)(smraw + (size_t)KDIM * NPIX * 2);  // INSZ floats, bn-applied

  const int b = blockIdx.x;
  const int tid = threadIdx.x;
  const int lane = tid & 31;
  const int wave = tid >> 5;
  const int lhalf = lane >> 4;
  const int lmod = lane & 15;

  // ---- phase 0: stage bn(h) into LDS ----
  const float* hb = hin + (size_t)b * INSZ;
  for (int i = tid * 16; i < INSZ; i += 128 * 16) __builtin_prefetch(hb + i, 0, 1);
  for (int i = tid; i < INSZ; i += 128) {
    int c = i / (HIN * WIN);
    hs[i] = hb[i] * sc[c] + sh[c];
  }
  __syncthreads();

  // ---- phase 1: deformable bilinear gather -> LDS f16 in B-fragment order ----
  const float* ob = off + (size_t)b * 18 * NPIX;
  for (int pos = tid; pos < 9 * NPIX; pos += 128) {
    int k = pos / NPIX;
    int pix = pos - k * NPIX;
    int ho = pix / WO, wo = pix - ho * WO;
    int ki = k / 3, kj = k - ki * 3;
    float dy = ob[(2 * k + 0) * NPIX + pix];
    float dx = ob[(2 * k + 1) * NPIX + pix];
    float py = (float)(ho * STRIDE - 1 + ki) + dy;
    float px = (float)(wo * STRIDE - 1 + kj) + dx;
    float y0f = floorf(py), x0f = floorf(px);
    float wy = py - y0f, wx = px - x0f;
    int y0 = (int)y0f, x0 = (int)x0f;
    int y1 = y0 + 1, x1 = x0 + 1;
    float my0 = ((unsigned)y0 < (unsigned)HIN) ? 1.0f : 0.0f;
    float my1 = ((unsigned)y1 < (unsigned)HIN) ? 1.0f : 0.0f;
    float mx0 = ((unsigned)x0 < (unsigned)WIN) ? 1.0f : 0.0f;
    float mx1 = ((unsigned)x1 < (unsigned)WIN) ? 1.0f : 0.0f;
    float w00 = (1.0f - wy) * (1.0f - wx) * my0 * mx0;
    float w01 = (1.0f - wy) * wx * my0 * mx1;
    float w10 = wy * (1.0f - wx) * my1 * mx0;
    float w11 = wy * wx * my1 * mx1;
    int y0c = min(max(y0, 0), HIN - 1), y1c = min(max(y1, 0), HIN - 1);
    int x0c = min(max(x0, 0), WIN - 1), x1c = min(max(x1, 0), WIN - 1);
    int l00 = y0c * WIN + x0c, l01 = y0c * WIN + x1c;
    int l10 = y1c * WIN + x0c, l11 = y1c * WIN + x1c;
    // destination slot pieces that depend only on pix
    const int nt = pix >> 4;
    const int lm = pix & 15;
    for (int ci = 0; ci < CIN; ++ci) {
      const float* hc = hs + ci * (HIN * WIN);
      float v = w00 * hc[l00] + w01 * hc[l01] + w10 * hc[l10] + w11 * hc[l11];
      int krow = ci * 9 + k;
      int kt = krow >> 5, kw = krow & 31;   // B layout: lanes 0-15 K={0..7,16..23}, 16-31 K={8..15,24..31}
      int addr = (((kt * NT + nt) << 5) + (((kw >> 3) & 1) << 4) + lm) * 16 + ((kw & 7) + ((kw & 16) >> 1));
      frag[addr] = (_Float16)v;
    }
  }
  __syncthreads();

  // ---- phase 2: WMMA GEMM ----
  const v16h* fb = (const v16h*)frag;
  const v16h* wa = (const v16h*)wfrag;
  for (int mt = wave; mt < NMT; mt += 4) {
    v8f acc[NT] = {};
    for (int kt = 0; kt < NKT; ++kt) {
      v16h a = wa[(size_t)(mt * NKT + kt) * 32 + lane];
#pragma unroll
      for (int nt = 0; nt < NT; ++nt) {
        v16h bf = fb[(kt * NT + nt) * 32 + lane];
        acc[nt] = __builtin_amdgcn_wmma_f32_16x16x32_f16(
            false, a, false, bf, (short)0, acc[nt], false, false);
      }
    }
    // ---- fused relu + store + per-channel stats ----
    float* obase = hout + ((size_t)b * COUT + mt * 16 + lhalf * 8) * NPIX + lmod;
    float rs[8], rq[8];
#pragma unroll
    for (int r = 0; r < 8; ++r) { rs[r] = 0.0f; rq[r] = 0.0f; }
#pragma unroll
    for (int nt = 0; nt < NT; ++nt) {
#pragma unroll
      for (int r = 0; r < 8; ++r) {
        float v = acc[nt][r];
        v = fmaxf(v, 0.0f);
        obase[r * NPIX + nt * 16] = v;
        rs[r] += v;
        rq[r] += v * v;
      }
    }
#pragma unroll
    for (int r = 0; r < 8; ++r) {
      float s = rs[r], q = rq[r];
      for (int msk = 8; msk >= 1; msk >>= 1) {
        s += __shfl_xor(s, msk);
        q += __shfl_xor(q, msk);
      }
      if (lmod == 0) {
        int cout = mt * 16 + lhalf * 8 + r;
        atomicAdd(&osum[cout], s);
        atomicAdd(&osumsq[cout], q);
      }
    }
  }
}

// ------------------------------------------------- avgpool(4x4) + bn4 + fc
__global__ void __launch_bounds__(128) pool_fc_kernel(const float* __restrict__ h4,
                                                      const float* __restrict__ sc,
                                                      const float* __restrict__ sh,
                                                      const float* __restrict__ fw,
                                                      const float* __restrict__ fb,
                                                      float* __restrict__ out) {
  __shared__ float p[128];
  const int b = blockIdx.x, tid = threadIdx.x;
  {
    const float* hb = h4 + ((size_t)b * 128 + tid) * 16;
    float s = 0.0f;
#pragma unroll
    for (int i = 0; i < 16; ++i) s += hb[i];
    p[tid] = (s * (1.0f / 16.0f)) * sc[tid] + sh[tid];
  }
  __syncthreads();
  if (tid < 10) {
    float acc = fb[tid];
    for (int c = 0; c < 128; ++c) acc += fw[tid * 128 + c] * p[c];
    out[(size_t)b * 10 + tid] = acc;
  }
}

// ---------------------------------------------------------------- launcher
extern "C" void kernel_launch(void* const* d_in, const int* in_sizes, int n_in,
                              void* d_out, int out_size, void* d_ws, size_t ws_size,
                              hipStream_t stream) {
  (void)in_sizes; (void)n_in; (void)out_size; (void)ws_size;
  const float* x       = (const float*)d_in[0];
  const float* conv1_w = (const float*)d_in[1];
  const float* conv1_b = (const float*)d_in[2];
  const float* bn1_g   = (const float*)d_in[3];
  const float* bn1_b   = (const float*)d_in[4];
  const float* off1_w  = (const float*)d_in[5];
  const float* off1_b  = (const float*)d_in[6];
  const float* dc1_w   = (const float*)d_in[7];
  const float* bn2_g   = (const float*)d_in[8];
  const float* bn2_b   = (const float*)d_in[9];
  const float* off2_w  = (const float*)d_in[10];
  const float* off2_b  = (const float*)d_in[11];
  const float* dc2_w   = (const float*)d_in[12];
  const float* bn3_g   = (const float*)d_in[13];
  const float* bn3_b   = (const float*)d_in[14];
  const float* off3_w  = (const float*)d_in[15];
  const float* off3_b  = (const float*)d_in[16];
  const float* dc3_w   = (const float*)d_in[17];
  const float* bn4_g   = (const float*)d_in[18];
  const float* bn4_b   = (const float*)d_in[19];
  const float* fc_w    = (const float*)d_in[20];
  const float* fc_b    = (const float*)d_in[21];
  float* out = (float*)d_out;

  // ---- workspace carve-up ----
  char* ws = (char*)d_ws;
  size_t cur = 0;
  auto alloc = [&](size_t bytes) -> char* {
    cur = (cur + 255) & ~(size_t)255;
    char* p = ws + cur;
    cur += bytes;
    return p;
  };
  float* h1   = (float*)alloc((size_t)NIMG * 32 * 256 * 4);
  float* h2   = (float*)alloc((size_t)NIMG * 64 * 64 * 4);
  float* h3   = (float*)alloc((size_t)NIMG * 128 * 64 * 4);
  float* h4   = (float*)alloc((size_t)NIMG * 128 * 16 * 4);
  float* off1 = (float*)alloc((size_t)NIMG * 18 * 64 * 4);
  float* off2 = (float*)alloc((size_t)NIMG * 18 * 64 * 4);
  float* off3 = (float*)alloc((size_t)NIMG * 18 * 16 * 4);
  _Float16* w1h = (_Float16*)alloc((size_t)64 * 288 * 2);
  _Float16* w2h = (_Float16*)alloc((size_t)128 * 576 * 2);
  _Float16* w3h = (_Float16*)alloc((size_t)128 * 1152 * 2);
  float* stats  = (float*)alloc(1408 * 4);
  // layout: L1[sum32,sq32,scale32,shift32] L2[4x64] L3[4x128] L4[4x128]
  float* s1sum = stats + 0,    *s1sq = stats + 32,   *s1sc = stats + 64,   *s1sh = stats + 96;
  float* s2sum = stats + 128,  *s2sq = stats + 192,  *s2sc = stats + 256,  *s2sh = stats + 320;
  float* s3sum = stats + 384,  *s3sq = stats + 512,  *s3sc = stats + 640,  *s3sh = stats + 768;
  float* s4sum = stats + 896,  *s4sq = stats + 1024, *s4sc = stats + 1152, *s4sh = stats + 1280;

  // stats must start at zero every call (BN atomics accumulate)
  zero_kernel<<<(1408 + 255) / 256, 256, 0, stream>>>(stats, 1408);

  // deform weights -> f16 in WMMA A-fragment order
  cvt_frag_kernel<<<(18432 + 255) / 256, 256, 0, stream>>>(dc1_w, w1h, 288, 18432);
  cvt_frag_kernel<<<(73728 + 255) / 256, 256, 0, stream>>>(dc2_w, w2h, 576, 73728);
  cvt_frag_kernel<<<(147456 + 255) / 256, 256, 0, stream>>>(dc3_w, w3h, 1152, 147456);

  // stage 1: conv1 + relu + bn1 stats -> bn1 finalize
  conv1_kernel<<<NIMG, 256, (324 + 64) * 4, stream>>>(x, conv1_w, conv1_b, h1, s1sum, s1sq);
  bn_finalize_kernel<<<1, 32, 0, stream>>>(s1sum, s1sq, bn1_g, bn1_b, s1sc, s1sh,
                                           1.0f / (2048.0f * 256.0f));

  // stage 2: off1 + deform conv 1 (32 -> 64, 16x16 -> 8x8, stride 2)
  off_conv_kernel<32, 16, 16, 8, 8, 2>
      <<<NIMG, 128, 32 * 256 * 4, stream>>>(h1, s1sc, s1sh, off1_w, off1_b, off1);
  deform_wmma_kernel<32, 64, 16, 16, 8, 8, 2>
      <<<NIMG, 128, 288 * 64 * 2 + 32 * 256 * 4, stream>>>(h1, s1sc, s1sh, off1, w1h,
                                                           h2, s2sum, s2sq);
  bn_finalize_kernel<<<1, 64, 0, stream>>>(s2sum, s2sq, bn2_g, bn2_b, s2sc, s2sh,
                                           1.0f / (2048.0f * 64.0f));

  // stage 3: off2 + deform conv 2 (64 -> 128, 8x8 -> 8x8, stride 1)
  off_conv_kernel<64, 8, 8, 8, 8, 1>
      <<<NIMG, 128, 64 * 64 * 4, stream>>>(h2, s2sc, s2sh, off2_w, off2_b, off2);
  deform_wmma_kernel<64, 128, 8, 8, 8, 8, 1>
      <<<NIMG, 128, 576 * 64 * 2 + 64 * 64 * 4, stream>>>(h2, s2sc, s2sh, off2, w2h,
                                                          h3, s3sum, s3sq);
  bn_finalize_kernel<<<1, 128, 0, stream>>>(s3sum, s3sq, bn3_g, bn3_b, s3sc, s3sh,
                                            1.0f / (2048.0f * 64.0f));

  // stage 4: off3 + deform conv 3 (128 -> 128, 8x8 -> 4x4, stride 2)
  off_conv_kernel<128, 8, 8, 4, 4, 2>
      <<<NIMG, 128, 128 * 64 * 4, stream>>>(h3, s3sc, s3sh, off3_w, off3_b, off3);
  deform_wmma_kernel<128, 128, 8, 8, 4, 4, 2>
      <<<NIMG, 128, 1152 * 16 * 2 + 128 * 64 * 4, stream>>>(h3, s3sc, s3sh, off3, w3h,
                                                            h4, s4sum, s4sq);
  bn_finalize_kernel<<<1, 128, 0, stream>>>(s4sum, s4sq, bn4_g, bn4_b, s4sc, s4sh,
                                            1.0f / (2048.0f * 16.0f));

  // stage 5: avgpool + bn4 + fc
  pool_fc_kernel<<<NIMG, 128, 0, stream>>>(h4, s4sc, s4sh, fc_w, fc_b, out);
}
